// TemporalAttention_3616362463686
// MI455X (gfx1250) — matrix operand order repaired and checked by reading, not compiled, not run on hardware
//
#include <hip/hip_runtime.h>
#include <hip/hip_bf16.h>

// ---------------------------------------------------------------------------
// Windowed causal multi-head attention, fused per 64-token window.
// B=8, T=8192, C=128, H=4, W=64, hd=32  -> 1024 windows, 1 workgroup each.
// Memory-bound (~67MB @ 23.3TB/s); all GEMMs on v_wmma_f32_16x16x32_f16.
// v is stored TRANSPOSED in LDS at creation so P@v B-fragments are
// contiguous ds_load_b128 instead of 16-way scalar gathers.
// ---------------------------------------------------------------------------

typedef _Float16 h16v __attribute__((ext_vector_type(16)));
typedef _Float16 h8v  __attribute__((ext_vector_type(8)));
typedef float    f8v  __attribute__((ext_vector_type(8)));
typedef float    f4v  __attribute__((ext_vector_type(4)));

#define WMMA16(a, b, c) \
  __builtin_amdgcn_wmma_f32_16x16x32_f16(false, (a), false, (b), (short)0, (c), false, false)

// A-fragment (16x32 f16) from LDS, row-major with given stride (halves).
// Lane m(0..15): K 0..15 contiguous;
// lane m+16:     K 16..31 contiguous.
__device__ __forceinline__ h16v load_A_lds(const _Float16* base, int stride, int lane) {
  const int mr = lane & 15, hf = lane >> 4;
  const _Float16* p0 = base + mr * stride + hf * 16;
  union { h16v v; h8v h[2]; } u;
  u.h[0] = *(const h8v*)p0;
  u.h[1] = *(const h8v*)(p0 + 8);
  return u.v;
}

// A-fragment from global f32 (convert to f16). Same layout as above.
__device__ __forceinline__ h16v load_A_gf32(const float* base, int stride, int kbase, int lane) {
  const int mr = lane & 15, hf = lane >> 4;
  const float* p0 = base + mr * stride + kbase + hf * 16;
  f4v a0 = *(const f4v*)(p0);
  f4v a1 = *(const f4v*)(p0 + 4);
  f4v b0 = *(const f4v*)(p0 + 8);
  f4v b1 = *(const f4v*)(p0 + 12);
  h16v r;
#pragma unroll
  for (int j = 0; j < 4; ++j) {
    r[j]      = (_Float16)a0[j];
    r[j + 4]  = (_Float16)a1[j];
    r[j + 8]  = (_Float16)b0[j];
    r[j + 12] = (_Float16)b1[j];
  }
  return r;
}

// B-fragment (32x16 f16) where B[k][n] = W[n][k], W row-major f16.
// Lane n(0..15): 16 consecutive K halves (K 0..15); lane n+16: K 16..31.
__device__ __forceinline__ h16v load_B_rows(const _Float16* base, int stride, int colbase, int lane) {
  const int nr = lane & 15, hf = lane >> 4;
  const _Float16* p = base + nr * stride + colbase + hf * 16;
  union { h16v v; h8v h[2]; } u;
  u.h[0] = *(const h8v*)p;
  u.h[1] = *(const h8v*)(p + 8);
  return u.v;
}

__global__ void wconv_kernel(const float* __restrict__ wqkv,
                             const float* __restrict__ wproj,
                             _Float16* __restrict__ dst) {
  const int i = blockIdx.x * 256 + threadIdx.x;
  const int NQ = 3 * 128 * 128;          // 49152
  const int NT = NQ + 128 * 128;         // 65536
  if (i < NQ)      dst[i] = (_Float16)wqkv[i];
  else if (i < NT) dst[i] = (_Float16)wproj[i - NQ];
}

__global__ __launch_bounds__(256)
void attn_win_kernel(const float* __restrict__ x,
                     const _Float16* __restrict__ wq,   // 384x128 f16
                     const _Float16* __restrict__ wp,   // 128x128 f16
                     const float* __restrict__ bias,    // 128
                     float* __restrict__ out) {
  // qkv buffer: 64 rows x 384 cols f16. q:0-127, k:128-255 row-major.
  // v-region (cols 256..383) holds v TRANSPOSED via flat map
  //   f = d*64 + n  ->  qkvh[(f>>7)*384 + 256 + (f&127)]   (d=0..127, n=0..63)
  // After attention, the q-region (cols 0..127) is reused for attn output.
  __shared__ __align__(16) _Float16 qkvh[64 * 384];
  // Per-wave P scratch: 8 waves x (16x64) f16. 16KB.
  __shared__ __align__(16) _Float16 ph[8 * 16 * 64];

  const int lane = threadIdx.x & 31;
  const int wave = threadIdx.x >> 5;
  const int mr = lane & 15;        // tile col (B/C/D) or tile row (A)
  const int hf = lane >> 4;        // lane half
  const int win = blockIdx.x;

  const float* xw = x + (size_t)win * 64 * 128;
  float*       ow = out + (size_t)win * 64 * 128;

  // -------- Stage 1: qkv = x @ wq^T  (64x384), 96 tiles, 12 per wave ------
  {
    const int mT = wave >> 1;                 // fixed 16-row strip per wave
    h16v a[4];
#pragma unroll
    for (int kc = 0; kc < 4; ++kc)
      a[kc] = load_A_gf32(xw + mT * 16 * 128, 128, kc * 32, lane);

    const int n0 = (wave & 1) * 12;
#pragma unroll
    for (int nt = 0; nt < 12; ++nt) {
      const int nT = n0 + nt;
      f8v c = {0.f, 0.f, 0.f, 0.f, 0.f, 0.f, 0.f, 0.f};
#pragma unroll
      for (int kc = 0; kc < 4; ++kc) {
        h16v b = load_B_rows(wq + nT * 16 * 128, 128, kc * 32, lane);
        c = WMMA16(a[kc], b, c);
      }
      if (nT < 16) {
        // q/k tiles: row-major scatter store (read back row-wise later)
        _Float16* dst = qkvh + (mT * 16 + hf * 8) * 384 + nT * 16 + mr;
#pragma unroll
        for (int r = 0; r < 8; ++r) dst[r * 384] = (_Float16)c[r];
      } else {
        // v tiles: transposed store; rows r are consecutive in f = d*64+m,
        // so one contiguous 16-byte ds_store_b128 per lane.
        const int dcol = (nT - 16) * 16 + mr;            // 0..127
        const int f0 = dcol * 64 + mT * 16 + hf * 8;     // multiple of 8
        _Float16* p = qkvh + (f0 >> 7) * 384 + 256 + (f0 & 127);
        h8v vh;
#pragma unroll
        for (int r = 0; r < 8; ++r) vh[r] = (_Float16)c[r];
        *(h8v*)p = vh;
      }
    }
  }
  __syncthreads();

  // -------- Stage 2: per-head attention. 2 waves per head, 2 row-tiles ----
  const int h = wave >> 1;
  const float scale = 0.17677669529663687f;  // 1/sqrt(32)
#pragma unroll 1
  for (int i = 0; i < 2; ++i) {
    const int mT = (wave & 1) * 2 + i;

    // S = q @ k^T  (16x64), hd=32 -> single K-chunk
    h16v aq = load_A_lds(qkvh + mT * 16 * 384 + h * 32, 384, lane);
    f8v s[4];
#pragma unroll
    for (int nT = 0; nT < 4; ++nT) {
      h16v bk = load_B_rows(qkvh + nT * 16 * 384 + 128 + h * 32, 384, 0, lane);
      f8v z = {0.f, 0.f, 0.f, 0.f, 0.f, 0.f, 0.f, 0.f};
      s[nT] = WMMA16(aq, bk, z);
    }

    // scale + causal mask + row softmax; P -> per-wave LDS scratch (f16)
#pragma unroll
    for (int r = 0; r < 8; ++r) {
      const int row = mT * 16 + hf * 8 + r;
      float mx = -3.0e38f;
#pragma unroll
      for (int nT = 0; nT < 4; ++nT) {
        const int col = nT * 16 + mr;
        float v = s[nT][r] * scale;
        v = (col <= row) ? v : -10000.0f;
        s[nT][r] = v;
        mx = fmaxf(mx, v);
      }
#pragma unroll
      for (int off = 1; off < 16; off <<= 1) mx = fmaxf(mx, __shfl_xor(mx, off, 32));
      float sum = 0.0f;
#pragma unroll
      for (int nT = 0; nT < 4; ++nT) {
        float e = __expf(s[nT][r] - mx);
        s[nT][r] = e;
        sum += e;
      }
#pragma unroll
      for (int off = 1; off < 16; off <<= 1) sum += __shfl_xor(sum, off, 32);
      const float inv = 1.0f / sum;
      _Float16* pdst = ph + wave * 1024 + (hf * 8 + r) * 64 + mr;
#pragma unroll
      for (int nT = 0; nT < 4; ++nT) pdst[nT * 16] = (_Float16)(s[nT][r] * inv);
    }

    __syncthreads();  // uniform: executed twice by every wave

    // O = P @ v  (16x32); A from P scratch, B from transposed v (contiguous)
    h16v ap[2];
    ap[0] = load_A_lds(ph + wave * 1024, 64, lane);
    ap[1] = load_A_lds(ph + wave * 1024 + 32, 64, lane);
#pragma unroll
    for (int dT = 0; dT < 2; ++dT) {
      f8v o = {0.f, 0.f, 0.f, 0.f, 0.f, 0.f, 0.f, 0.f};
#pragma unroll
      for (int kc = 0; kc < 2; ++kc) {
        // B[k=n][col=d]: lane col d = h*32+dT*16+mr; K run = kc*32+hf*16+0..15
        // -> 16 consecutive halves in the transposed-v flat map (no 128-fold
        //    crossing since (d*64 + kbase) mod 128 <= 112).
        const int dcol = h * 32 + dT * 16 + mr;
        const int f0 = dcol * 64 + kc * 32 + hf * 16;
        const _Float16* p = qkvh + (f0 >> 7) * 384 + 256 + (f0 & 127);
        union { h16v v; h8v h[2]; } u;
        u.h[0] = *(const h8v*)p;
        u.h[1] = *(const h8v*)(p + 8);
        o = WMMA16(ap[kc], u.v, o);
      }
      // write attention output into dead q-region (cols 0..127), stride 384
      _Float16* adst = qkvh + (mT * 16 + hf * 8) * 384 + h * 32 + dT * 16 + mr;
#pragma unroll
      for (int r = 0; r < 8; ++r) adst[r * 384] = (_Float16)o[r];
    }
  }
  __syncthreads();

  // -------- Stage 3: out = att @ wp^T + b  (64x128), 32 tiles, 4/wave ----
  {
    const int mT = wave >> 1;
    h16v a[4];
#pragma unroll
    for (int kc = 0; kc < 4; ++kc)
      a[kc] = load_A_lds(qkvh + mT * 16 * 384 + kc * 32, 384, lane);

    const int n0 = (wave & 1) * 4;
#pragma unroll
    for (int nt = 0; nt < 4; ++nt) {
      const int nT = n0 + nt;
      f8v c = {0.f, 0.f, 0.f, 0.f, 0.f, 0.f, 0.f, 0.f};
#pragma unroll
      for (int kc = 0; kc < 4; ++kc) {
        h16v b = load_B_rows(wp + nT * 16 * 128, 128, kc * 32, lane);
        c = WMMA16(a[kc], b, c);
      }
      const float bv = bias[nT * 16 + mr];
      float* dst = ow + (mT * 16 + hf * 8) * 128 + nT * 16 + mr;
#pragma unroll
      for (int r = 0; r < 8; ++r) dst[r * 128] = c[r] + bv;
    }
  }
}

extern "C" void kernel_launch(void* const* d_in, const int* in_sizes, int n_in,
                              void* d_out, int out_size, void* d_ws, size_t ws_size,
                              hipStream_t stream) {
  const float* x     = (const float*)d_in[0];
  const float* wqkv  = (const float*)d_in[1];
  const float* wproj = (const float*)d_in[2];
  const float* bproj = (const float*)d_in[3];
  float* out = (float*)d_out;

  _Float16* wh = (_Float16*)d_ws;             // 65536 halves = 128KB scratch

  // Convert weights to f16 once; stays resident in L2 across all windows.
  wconv_kernel<<<256, 256, 0, stream>>>(wqkv, wproj, wh);

  const int nwin = in_sizes[0] / (64 * 128);  // 1024 windows
  attn_win_kernel<<<nwin, 256, 0, stream>>>(x, wh, wh + 3 * 128 * 128, bproj, out);
}
